// MultiHeadSelfAttention_10557029614217
// MI455X (gfx1250) — matrix operand order repaired and checked by reading, not compiled, not run on hardware
//
#include <hip/hip_runtime.h>

typedef __attribute__((ext_vector_type(16))) _Float16 v16h;
typedef __attribute__((ext_vector_type(8)))  _Float16 v8h;
typedef __attribute__((ext_vector_type(8)))  float    v8f;

// Problem dims
#define Bd 2
#define Sd 2048
#define Ed 1024
#define Hd 16
#define Dd 64
#define HEAD_ELEMS (Sd * Dd)                  // 131072
#define QKV_ELEMS  (3 * Bd * Hd * Sd * Dd)    // 12582912 f16
#define ATT_ELEMS  (Bd * Sd * Ed)             // 4194304 f16

static __device__ __forceinline__ v8f wmma_f16(v16h a, v16h b, v8f c) {
  // D = A(16x32 f16) * B(32x16 f16) + C(16x16 f32)
  return __builtin_amdgcn_wmma_f32_16x16x32_f16(false, a, false, b, (short)0, c,
                                                false, false);
}

// ---- CDNA5 async global->LDS copy (ASYNCcnt-tracked, no VGPR roundtrip) ----
typedef __attribute__((address_space(3))) _Float16 lds_f16;

static __device__ __forceinline__ void async_copy_b128(const _Float16* gsrc,
                                                       _Float16* ldst) {
  const unsigned lds_off =
      (unsigned)(unsigned long long)(lds_f16*)ldst;       // raw LDS byte offset
  const unsigned long long gaddr = (unsigned long long)gsrc;
  asm volatile("global_load_async_to_lds_b128 %0, %1, off"
               :: "v"(lds_off), "v"(gaddr)
               : "memory");
}

static __device__ __forceinline__ void async_wait0() {
#if __has_builtin(__builtin_amdgcn_s_wait_asynccnt)
  __builtin_amdgcn_s_wait_asynccnt(0);
#else
  asm volatile("s_wait_asynccnt 0x0" ::: "memory");
#endif
}

// A-fragment (16x32 f16) from LDS, row-major [16][stride] halves.
// Contiguous k order (same as B): lane<16 holds K0..15, lane>=16 holds K16..31.
static __device__ __forceinline__ v16h ldsA_frag(const _Float16* base, int stride, int lane) {
  const int row = lane & 15;
  const int c0  = (lane < 16) ? 0 : 16;
  const v8h lo = *(const v8h*)(base + row * stride + c0);
  const v8h hi = *(const v8h*)(base + row * stride + c0 + 8);
  v16h a;
#pragma unroll
  for (int e = 0; e < 8; e++) { a[e] = lo[e]; a[e + 8] = hi[e]; }
  return a;
}

// B-fragment (32x16 f16) from LDS stored K-contiguous per N: Bt[n][k].
// Contiguous k order (same as A): lane%16 = N; lane<16 holds K0..15, lane>=16 holds K16..31.
static __device__ __forceinline__ v16h ldsB_frag(const _Float16* base, int stride, int lane) {
  const int n  = lane & 15;
  const int kb = (lane < 16) ? 0 : 16;
  const v8h lo = *(const v8h*)(base + n * stride + kb);
  const v8h hi = *(const v8h*)(base + n * stride + kb + 8);
  v16h b;
#pragma unroll
  for (int e = 0; e < 8; e++) { b[e] = lo[e]; b[e + 8] = hi[e]; }
  return b;
}

// ---------------------------------------------------------------------------
// Kernel 1: QKV projection.  C[4096,3072] = X[4096,1024] @ Wqkv[1024,3072]+b
// Output scattered as f16 into head-major [which][b][h][s][d].
// (f32 inputs must convert -> VGPR staging path)
// ---------------------------------------------------------------------------
__global__ __launch_bounds__(256)
void qkv_gemm_kernel(const float* __restrict__ x, const float* __restrict__ w,
                     const float* __restrict__ bias, _Float16* __restrict__ qkv) {
  __shared__ _Float16 As[128 * 40];   // [m][k] k-stride 40 (pad)
  __shared__ _Float16 Bs[64 * 40];    // [n][k] transposed, k-contiguous

  const int tid  = threadIdx.x;
  const int lane = tid & 31;
  const int wv   = tid >> 5;
  const int mBase = blockIdx.y * 128;
  const int nBase = blockIdx.x * 64;
  const int wm = (wv & 3) * 32;
  const int wn = (wv >> 2) * 32;

  v8f acc[2][2] = {};

  for (int k0 = 0; k0 < Ed; k0 += 32) {
    __syncthreads();
    {  // A tile 128x32: f32 -> f16
      const int row = tid >> 1, ch = (tid & 1) * 16;
      const float* src = x + (size_t)(mBase + row) * Ed + k0 + ch;
      if (k0 + 32 < Ed) __builtin_prefetch(src + 32, 0, 0);
      v8h t0, t1;
#pragma unroll
      for (int e = 0; e < 8; e++) t0[e] = (_Float16)src[e];
#pragma unroll
      for (int e = 0; e < 8; e++) t1[e] = (_Float16)src[8 + e];
      *(v8h*)&As[row * 40 + ch]     = t0;
      *(v8h*)&As[row * 40 + ch + 8] = t1;
    }
    {  // B tile 32x64 -> Bs[n][k] (transpose), f32 -> f16
      const int kr = tid >> 3, c0 = (tid & 7) * 8;
      const float* src = w + (size_t)(k0 + kr) * 3072 + nBase + c0;
#pragma unroll
      for (int e = 0; e < 8; e++) Bs[(c0 + e) * 40 + kr] = (_Float16)src[e];
    }
    __syncthreads();

    v16h af[2], bf[2];
#pragma unroll
    for (int i = 0; i < 2; i++) af[i] = ldsA_frag(&As[(wm + i * 16) * 40], 40, lane);
#pragma unroll
    for (int j = 0; j < 2; j++) bf[j] = ldsB_frag(&Bs[(wn + j * 16) * 40], 40, lane);
#pragma unroll
    for (int i = 0; i < 2; i++)
#pragma unroll
      for (int j = 0; j < 2; j++) acc[i][j] = wmma_f16(af[i], bf[j], acc[i][j]);
  }

  // Writeback: C layout -> [which][b][h][s][d] f16 (+bias)
#pragma unroll
  for (int j = 0; j < 2; j++) {
    const int n     = nBase + wn + j * 16 + (lane & 15);
    const float bv  = bias[n];
    const int which = n >> 10;
    const int rem   = n & 1023;
    const int h     = rem >> 6;
    const int d     = rem & 63;
#pragma unroll
    for (int i = 0; i < 2; i++) {
#pragma unroll
      for (int r = 0; r < 8; r++) {
        const int m    = mBase + wm + i * 16 + r + ((lane >> 4) << 3);
        const int bIdx = m >> 11;           // / S
        const int s    = m & (Sd - 1);
        const size_t off = (((size_t)which * Bd + bIdx) * Hd + h) * HEAD_ELEMS +
                           (size_t)s * Dd + d;
        qkv[off] = (_Float16)(acc[i][j][r] + bv);
      }
    }
  }
}

// ---------------------------------------------------------------------------
// Kernel 2: causal flash attention per (b,h).  Q-tile 64 rows / block,
// 4 waves x 16 rows; 32-key blocks staged in LDS (K via async-to-LDS DMA);
// online softmax.
// ---------------------------------------------------------------------------
__global__ __launch_bounds__(128)
void attn_kernel(const _Float16* __restrict__ qkv, _Float16* __restrict__ attnOut) {
  __shared__ _Float16 Ks[32 * 72];      // [key][d]  (B-frag: n=key, k=d contiguous)
  __shared__ _Float16 Vt[64 * 40];      // [d][key]  (B-frag: n=d, k=key contiguous)
  __shared__ _Float16 Ps[4][16 * 40];   // per-wave P tile [q][key]

  const int tid  = threadIdx.x;
  const int lane = tid & 31;
  const int wv   = tid >> 5;
  const int qb   = blockIdx.x * 64;
  const int bh   = blockIdx.y;          // b*H + h
  const size_t headOff = (size_t)bh * HEAD_ELEMS;
  const _Float16* Q = qkv + headOff;
  const _Float16* K = qkv + (size_t)1 * Bd * Hd * HEAD_ELEMS + headOff;
  const _Float16* V = qkv + (size_t)2 * Bd * Hd * HEAD_ELEMS + headOff;

  // Q A-fragments for the two 32-wide d k-steps (held for whole kernel)
  const int qrow = qb + wv * 16 + (lane & 15);
  v16h qf[2];
#pragma unroll
  for (int kk = 0; kk < 2; kk++) {
    const int c0 = kk * 32 + ((lane < 16) ? 0 : 16);
    const v8h lo = *(const v8h*)(Q + (size_t)qrow * Dd + c0);
    const v8h hi = *(const v8h*)(Q + (size_t)qrow * Dd + c0 + 8);
#pragma unroll
    for (int e = 0; e < 8; e++) { qf[kk][e] = lo[e]; qf[kk][e + 8] = hi[e]; }
  }

  v8f oacc[4] = {};
  float mrun[8], lrun[8];
#pragma unroll
  for (int r = 0; r < 8; r++) { mrun[r] = -1e30f; lrun[r] = 0.0f; }

  const int nKB = (qb >> 5) + 2;        // causal: key blocks with base <= qb+63
  for (int kbIdx = 0; kbIdx < nKB; kbIdx++) {
    const int kbase = kbIdx * 32;
    __syncthreads();
    {  // K block [32][64] -> Ks : async global->LDS DMA (no VGPR roundtrip)
      const int key = tid >> 2, ch = (tid & 3) * 16;
      const _Float16* src = K + (size_t)(kbase + key) * Dd + ch;
      async_copy_b128(src,     &Ks[key * 72 + ch]);
      async_copy_b128(src + 8, &Ks[key * 72 + ch + 8]);
    }
    {  // V block transposed -> Vt[d][key] (needs transpose: VGPR path)
      const int key = tid >> 2, ch = (tid & 3) * 16;
      const _Float16* src = V + (size_t)(kbase + key) * Dd + ch;
      const v8h a = *(const v8h*)src;
      const v8h b = *(const v8h*)(src + 8);
#pragma unroll
      for (int e = 0; e < 8; e++) Vt[(ch + e) * 40 + key]     = a[e];
#pragma unroll
      for (int e = 0; e < 8; e++) Vt[(ch + 8 + e) * 40 + key] = b[e];
    }
    async_wait0();          // K DMA complete (ASYNCcnt==0) before barrier
    __syncthreads();

    // Scores: 2 key subtiles x 2 d k-steps
    v8f sc[2] = {};
#pragma unroll
    for (int j = 0; j < 2; j++) {
#pragma unroll
      for (int kk = 0; kk < 2; kk++) {
        const int n     = j * 16 + (lane & 15);
        const int kboff = kk * 32 + ((lane < 16) ? 0 : 16);
        const v8h lo = *(const v8h*)&Ks[n * 72 + kboff];
        const v8h hi = *(const v8h*)&Ks[n * 72 + kboff + 8];
        v16h bf;
#pragma unroll
        for (int e = 0; e < 8; e++) { bf[e] = lo[e]; bf[e + 8] = hi[e]; }
        sc[j] = wmma_f16(qf[kk], bf, sc[j]);
      }
    }

    // Mask, scale, online softmax (row r lives in C-vgpr r, per 16-lane half)
    const int colKey = kbase + (lane & 15);
    const int rowQ   = qb + wv * 16 + ((lane >> 4) << 3);
    float pnew[2][8];
#pragma unroll
    for (int r = 0; r < 8; r++) {
      float s0 = sc[0][r] * 0.125f;     // 1/sqrt(64)
      float s1 = sc[1][r] * 0.125f;
      const int q = rowQ + r;
      if (colKey > q)      s0 = -1e30f;
      if (colKey + 16 > q) s1 = -1e30f;
      float mx = fmaxf(s0, s1);
#pragma unroll
      for (int off = 1; off < 16; off <<= 1) mx = fmaxf(mx, __shfl_xor(mx, off, 32));
      const float mnew  = fmaxf(mrun[r], mx);
      const float alpha = __expf(mrun[r] - mnew);
      const float p0 = __expf(s0 - mnew);
      const float p1 = __expf(s1 - mnew);
      float rs = p0 + p1;
#pragma unroll
      for (int off = 1; off < 16; off <<= 1) rs += __shfl_xor(rs, off, 32);
      lrun[r] = lrun[r] * alpha + rs;
      mrun[r] = mnew;
#pragma unroll
      for (int jo = 0; jo < 4; jo++) oacc[jo][r] *= alpha;
      pnew[0][r] = p0;
      pnew[1][r] = p1;
    }

    // P (C layout) -> wave-private LDS -> A-fragment layout
    _Float16* pw = Ps[wv];
#pragma unroll
    for (int j = 0; j < 2; j++)
#pragma unroll
      for (int r = 0; r < 8; r++)
        pw[(r + ((lane >> 4) << 3)) * 40 + j * 16 + (lane & 15)] = (_Float16)pnew[j][r];

    v16h pf;
    {
      const int row = lane & 15;
      const int c0  = (lane < 16) ? 0 : 16;
      const v8h lo = *(const v8h*)(pw + row * 40 + c0);
      const v8h hi = *(const v8h*)(pw + row * 40 + c0 + 8);
#pragma unroll
      for (int e = 0; e < 8; e++) { pf[e] = lo[e]; pf[e + 8] = hi[e]; }
    }

    // O += P(16x32) @ V(32x64): 4 d-subtiles
#pragma unroll
    for (int jo = 0; jo < 4; jo++) {
      const int n     = jo * 16 + (lane & 15);
      const int kboff = (lane < 16) ? 0 : 16;
      const v8h lo = *(const v8h*)&Vt[n * 40 + kboff];
      const v8h hi = *(const v8h*)&Vt[n * 40 + kboff + 8];
      v16h vf;
#pragma unroll
      for (int e = 0; e < 8; e++) { vf[e] = lo[e]; vf[e + 8] = hi[e]; }
      oacc[jo] = wmma_f16(pf, vf, oacc[jo]);
    }
  }

  // Epilogue: O /= l, write f16 attn output [b][s][h*64+d]
  const int bIdx = bh >> 4;
  const int h    = bh & 15;
#pragma unroll
  for (int jo = 0; jo < 4; jo++) {
    const int d = jo * 16 + (lane & 15);
#pragma unroll
    for (int r = 0; r < 8; r++) {
      const int s = qb + wv * 16 + r + ((lane >> 4) << 3);
      const float val = oacc[jo][r] / lrun[r];
      attnOut[((size_t)bIdx * Sd + s) * Ed + h * Dd + d] = (_Float16)val;
    }
  }
}

// ---------------------------------------------------------------------------
// Kernel 3: output projection. out[4096,1024] = attn(f16) @ Wout(f32) + b
// A tile is pure f16 copy -> async global->LDS DMA.
// ---------------------------------------------------------------------------
__global__ __launch_bounds__(256)
void out_gemm_kernel(const _Float16* __restrict__ a, const float* __restrict__ w,
                     const float* __restrict__ bias, float* __restrict__ out) {
  __shared__ _Float16 As[128 * 40];
  __shared__ _Float16 Bs[64 * 40];

  const int tid  = threadIdx.x;
  const int lane = tid & 31;
  const int wv   = tid >> 5;
  const int mBase = blockIdx.y * 128;
  const int nBase = blockIdx.x * 64;
  const int wm = (wv & 3) * 32;
  const int wn = (wv >> 2) * 32;

  v8f acc[2][2] = {};

  for (int k0 = 0; k0 < Ed; k0 += 32) {
    __syncthreads();
    {  // A tile (already f16): async global->LDS DMA
      const int row = tid >> 1, ch = (tid & 1) * 16;
      const _Float16* src = a + (size_t)(mBase + row) * Ed + k0 + ch;
      async_copy_b128(src,     &As[row * 40 + ch]);
      async_copy_b128(src + 8, &As[row * 40 + ch + 8]);
    }
    {  // B tile f32 -> f16, transposed (conversion: VGPR path)
      const int kr = tid >> 3, c0 = (tid & 7) * 8;
      const float* src = w + (size_t)(k0 + kr) * Ed + nBase + c0;
      if (k0 + 32 < Ed) __builtin_prefetch(src + (size_t)32 * Ed, 0, 0);
#pragma unroll
      for (int e = 0; e < 8; e++) Bs[(c0 + e) * 40 + kr] = (_Float16)src[e];
    }
    async_wait0();
    __syncthreads();

    v16h af[2], bf[2];
#pragma unroll
    for (int i = 0; i < 2; i++) af[i] = ldsA_frag(&As[(wm + i * 16) * 40], 40, lane);
#pragma unroll
    for (int j = 0; j < 2; j++) bf[j] = ldsB_frag(&Bs[(wn + j * 16) * 40], 40, lane);
#pragma unroll
    for (int i = 0; i < 2; i++)
#pragma unroll
      for (int j = 0; j < 2; j++) acc[i][j] = wmma_f16(af[i], bf[j], acc[i][j]);
  }

#pragma unroll
  for (int j = 0; j < 2; j++) {
    const int n    = nBase + wn + j * 16 + (lane & 15);
    const float bv = bias[n];
#pragma unroll
    for (int i = 0; i < 2; i++) {
#pragma unroll
      for (int r = 0; r < 8; r++) {
        const int m = mBase + wm + i * 16 + r + ((lane >> 4) << 3);
        out[(size_t)m * Ed + n] = acc[i][j][r] + bv;
      }
    }
  }
}

// ---------------------------------------------------------------------------
extern "C" void kernel_launch(void* const* d_in, const int* in_sizes, int n_in,
                              void* d_out, int out_size, void* d_ws, size_t ws_size,
                              hipStream_t stream) {
  const float* x     = (const float*)d_in[0];
  const float* w_qkv = (const float*)d_in[1];
  const float* b_qkv = (const float*)d_in[2];
  const float* w_out = (const float*)d_in[3];
  const float* b_out = (const float*)d_in[4];

  _Float16* qkv  = (_Float16*)d_ws;                 // 3*B*H*S*D f16 (24 MB)
  _Float16* attn = qkv + QKV_ELEMS;                 // B*S*E f16 (8 MB)
  float* out = (float*)d_out;

  // 1) QKV projection: [4096,1024] x [1024,3072]
  hipLaunchKernelGGL(qkv_gemm_kernel, dim3(3072 / 64, 4096 / 128), dim3(256), 0,
                     stream, x, w_qkv, b_qkv, qkv);
  // 2) Causal flash attention per (b,h)
  hipLaunchKernelGGL(attn_kernel, dim3(Sd / 64, Bd * Hd), dim3(128), 0,
                     stream, qkv, attn);
  // 3) Output projection: [4096,1024] x [1024,1024]
  hipLaunchKernelGGL(out_gemm_kernel, dim3(Ed / 64, 4096 / 128), dim3(256), 0,
                     stream, attn, w_out, b_out, out);
}
